// TransformerEncoder_47090021434045
// MI455X (gfx1250) — matrix operand order repaired and checked by reading, not compiled, not run on hardware
//
#include <hip/hip_runtime.h>
#include <hip/hip_bf16.h>
#include <math.h>

// ---- problem constants (match reference) ----
#define BB 2
#define SS 2048
#define DD 1024
#define HH 16
#define FF 4096
#define LL 3
#define MM (BB*SS)      // 4096 token rows
#define DK 64

typedef __attribute__((ext_vector_type(16))) __bf16 v16bf;
typedef __attribute__((ext_vector_type(8)))  float  v8f;
typedef int v4i __attribute__((vector_size(16)));   // matches builtin param type

// ---- CDNA5 async global->LDS staging (ASYNCcnt-tracked DMA), with fallback ----
#if defined(__has_builtin)
# if __has_builtin(__builtin_amdgcn_global_load_async_to_lds_b128)
#  define HAVE_ASYNC_LDS 1
# endif
#endif
#ifndef HAVE_ASYNC_LDS
# define HAVE_ASYNC_LDS 0
#endif

#if HAVE_ASYNC_LDS
# define GLOBAL_TO_LDS_B128(gsrc, lds)                                          \
    __builtin_amdgcn_global_load_async_to_lds_b128(                             \
        (__attribute__((address_space(1))) v4i*)(gsrc),                         \
        (__attribute__((address_space(3))) v4i*)(lds), 0, 0)
# if __has_builtin(__builtin_amdgcn_s_wait_asynccnt)
#  define WAIT_ASYNC() __builtin_amdgcn_s_wait_asynccnt(0)
# else
#  define WAIT_ASYNC() asm volatile("s_wait_asynccnt 0x0" ::: "memory")
# endif
#else
# define GLOBAL_TO_LDS_B128(gsrc, lds)                                          \
    (*reinterpret_cast<uint4*>(lds) = *reinterpret_cast<const uint4*>(gsrc))
# define WAIT_ASYNC() ((void)0)
#endif

// ---- helpers ----
static __device__ inline __bf16 f2bf(float f) {
  unsigned u = __builtin_bit_cast(unsigned, f);
  u += 0x7fffu + ((u >> 16) & 1u);
  unsigned short h = (unsigned short)(u >> 16);
  return __builtin_bit_cast(__bf16, h);
}

static __device__ inline v8f wmma_bf16(v16bf a, v16bf b, v8f c) {
  return __builtin_amdgcn_wmma_f32_16x16x32_bf16(false, a, false, b, (short)0, c, false, false);
}

// build a 16-element bf16 fragment from two contiguous 16-byte chunks
union UFrag { uint4 q[2]; v16bf v; };
static __device__ inline v16bf ld_frag(const __bf16* p0, const __bf16* p1) {
  UFrag u;
  u.q[0] = *reinterpret_cast<const uint4*>(p0);
  u.q[1] = *reinterpret_cast<const uint4*>(p1);
  return u.v;
}

static __device__ inline float rmax16(float v) {
  #pragma unroll
  for (int m = 8; m >= 1; m >>= 1) v = fmaxf(v, __shfl_xor(v, m, 16));
  return v;
}
static __device__ inline float rsum16(float v) {
  #pragma unroll
  for (int m = 8; m >= 1; m >>= 1) v += __shfl_xor(v, m, 16);
  return v;
}

// ---- weight f32 -> bf16 conversion ----
__global__ __launch_bounds__(256) void f32_to_bf16_k(const float* __restrict__ s,
                                                     __bf16* __restrict__ d, int n) {
  int i = blockIdx.x * 256 + threadIdx.x;
  if (i < n) d[i] = f2bf(s[i]);
}

// ---- embedding + sinusoidal PE: x = emb[tok]*sqrt(D) + pe ----
__global__ __launch_bounds__(256) void embed_k(const int* __restrict__ tok,
                                               const float* __restrict__ emb,
                                               const float* __restrict__ pe,
                                               float* __restrict__ x) {
  int row = blockIdx.x;               // 0..MM-1
  int s = row % SS;
  int t = tok[row];
  const float* e = emb + (size_t)t * DD;
  const float* p = pe + (size_t)s * DD;
  float* o = x + (size_t)row * DD;
  for (int c = threadIdx.x; c < DD; c += 256) o[c] = e[c] * 32.0f + p[c]; // sqrt(1024)=32
}

// ---- layernorm (Bessel std, eps on std).  OUT==0 -> bf16, OUT==1 -> f32 ----
template<int OUT>
__global__ __launch_bounds__(256) void ln_k(const float* __restrict__ x,
                                            const float* __restrict__ ga,
                                            const float* __restrict__ gb,
                                            __bf16* __restrict__ ob,
                                            float* __restrict__ of) {
  __shared__ float red[256];
  int row = blockIdx.x;
  const float* xr = x + (size_t)row * DD;
  float s = 0.f;
  for (int c = threadIdx.x; c < DD; c += 256) s += xr[c];
  red[threadIdx.x] = s; __syncthreads();
  for (int st = 128; st > 0; st >>= 1) {
    if (threadIdx.x < st) red[threadIdx.x] += red[threadIdx.x + st];
    __syncthreads();
  }
  float mean = red[0] * (1.0f / DD);
  __syncthreads();
  float s2 = 0.f;
  for (int c = threadIdx.x; c < DD; c += 256) { float d = xr[c] - mean; s2 += d * d; }
  red[threadIdx.x] = s2; __syncthreads();
  for (int st = 128; st > 0; st >>= 1) {
    if (threadIdx.x < st) red[threadIdx.x] += red[threadIdx.x + st];
    __syncthreads();
  }
  float inv = 1.0f / (sqrtf(red[0] / (float)(DD - 1)) + 1e-6f);
  for (int c = threadIdx.x; c < DD; c += 256) {
    float v = ga[c] * (xr[c] - mean) * inv + gb[c];
    if (OUT == 0) ob[(size_t)row * DD + c] = f2bf(v);
    else          of[(size_t)row * DD + c] = v;
  }
}

// ---- tiled WMMA bf16 GEMM: C[M,N] = A[M,K] @ W[K,N] + bias, epilogue:
//   EPI 0: bf16 out; EPI 1: relu -> bf16 out; EPI 2: resid f32 += acc+bias ----
template<int EPI>
__global__ __launch_bounds__(256) void gemm_bf16_k(const __bf16* __restrict__ A,
                                                   const __bf16* __restrict__ Bw,
                                                   const float* __restrict__ bias,
                                                   __bf16* __restrict__ outb,
                                                   float* __restrict__ resid,
                                                   int M, int N, int K) {
  __shared__ __align__(16) __bf16 As[128][40];     // 128 rows x 32 k, +8 pad
  __shared__ __align__(16) __bf16 Bs2[128][40];    // TRANSPOSED: 128 cols x 32 k, +8 pad
  int tid = threadIdx.x;
  int lane = tid & 31, wave = tid >> 5;
  int ll = lane & 15, lh = lane >> 4;
  int wm = wave >> 2, wn = wave & 3;               // 2 x 4 wave grid
  int bm = blockIdx.x * 128;
  int bn = blockIdx.y * 128;
  v8f acc[4][2] = {};

  for (int k0 = 0; k0 < K; k0 += 32) {
    // stage A tile (128 rows x 32 k): async DMA straight into LDS
    #pragma unroll
    for (int c = tid; c < 512; c += 256) {
      int r = c >> 2, k8 = (c & 3) << 3;
      const __bf16* src = A + (size_t)(bm + r) * K + k0 + k8;
      GLOBAL_TO_LDS_B128(src, &As[r][k8]);
    }
    // stage B tile (32 k x 128 cols) transposed into Bs2[col][k]
    #pragma unroll
    for (int c = tid; c < 512; c += 256) {
      int r = c >> 4, n8 = (c & 15) << 3;
      const __bf16* src = Bw + (size_t)(k0 + r) * N + bn + n8;
      union { uint4 q; __bf16 h[8]; } u;
      u.q = *reinterpret_cast<const uint4*>(src);
      if (k0 + 32 < K) __builtin_prefetch(src + 32, 0, 0);   // global_prefetch_b8
      #pragma unroll
      for (int i = 0; i < 8; ++i) Bs2[n8 + i][r] = u.h[i];
    }
    WAIT_ASYNC();
    __syncthreads();

    v16bf af[4], bf_[2];
    #pragma unroll
    for (int mi = 0; mi < 4; ++mi) {
      int row = wm * 64 + mi * 16 + ll;
      af[mi] = ld_frag(&As[row][lh * 8], &As[row][16 + lh * 8]);  // K halves per lane group
    }
    #pragma unroll
    for (int ni = 0; ni < 2; ++ni) {
      int col = wn * 32 + ni * 16 + ll;
      bf_[ni] = ld_frag(&Bs2[col][lh * 8], &Bs2[col][16 + lh * 8]);  // K halves per lane group
    }
    #pragma unroll
    for (int mi = 0; mi < 4; ++mi)
      #pragma unroll
      for (int ni = 0; ni < 2; ++ni)
        acc[mi][ni] = wmma_bf16(af[mi], bf_[ni], acc[mi][ni]);
    __syncthreads();
  }

  // epilogue (C layout: vgpr r -> row r + 8*lh, col = ll within 16-tile)
  #pragma unroll
  for (int mi = 0; mi < 4; ++mi) {
    #pragma unroll
    for (int ni = 0; ni < 2; ++ni) {
      int col = bn + wn * 32 + ni * 16 + ll;
      float bv = bias ? bias[col] : 0.f;
      #pragma unroll
      for (int r = 0; r < 8; ++r) {
        int row = bm + wm * 64 + mi * 16 + r + lh * 8;
        float v = acc[mi][ni][r] + bv;
        size_t idx = (size_t)row * N + col;
        if (EPI == 0)      outb[idx] = f2bf(v);
        else if (EPI == 1) outb[idx] = f2bf(v > 0.f ? v : 0.f);
        else               resid[idx] += v;
      }
    }
  }
}

// ---- flash attention: 4 waves/block share (batch, head, key tiles);
//      each wave owns a 16-query tile; 32 keys per iteration; WMMA both GEMMs ----
__global__ __launch_bounds__(128) void attn_k(const __bf16* __restrict__ qb,
                                              const __bf16* __restrict__ kb,
                                              const __bf16* __restrict__ vb,
                                              const int* __restrict__ maskp,
                                              __bf16* __restrict__ ob) {
  __shared__ __align__(16) __bf16 Ks[32][72];       // key rows x 64 dims (+8 pad)
  __shared__ __align__(16) __bf16 Vt[64][40];       // TRANSPOSED: dim rows x 32 keys (+8 pad)
  __shared__ __align__(16) __bf16 ps[4][16 * 40];   // per-wave P staging (16 x 32, pad)
  int tid = threadIdx.x;
  int lane = tid & 31, wave = tid >> 5;
  int ll = lane & 15, lh = lane >> 4;
  int h = blockIdx.y, bz = blockIdx.z;
  int q0 = blockIdx.x * 64 + wave * 16;
  size_t rowbase = (size_t)bz * SS;
  int hb = h * DK;

  // Q fragments (16 x 64 -> two 16x32 A-frags), contiguous 16B chunks per lane
  const __bf16* qr = qb + (rowbase + q0 + ll) * DD + hb;
  v16bf a0 = ld_frag(qr + lh * 8, qr + 16 + lh * 8);
  v16bf a1 = ld_frag(qr + 32 + lh * 8, qr + 48 + lh * 8);

  float rm[8], rl[8];
  #pragma unroll
  for (int r = 0; r < 8; ++r) { rm[r] = -1e30f; rl[r] = 0.f; }
  v8f acc[4] = {};
  const int* mrow = maskp + bz * SS;
  __bf16* pw = ps[wave];

  for (int j = 0; j < SS / 32; ++j) {
    int kbase = j * 32;
    __syncthreads();   // previous-iteration LDS reads complete
    // stage K tile [32 keys][64 dims]: async DMA straight into LDS
    #pragma unroll
    for (int c = tid; c < 256; c += 128) {
      int key = c >> 3, d8 = (c & 7) << 3;
      const __bf16* src = kb + (rowbase + kbase + key) * DD + hb + d8;
      GLOBAL_TO_LDS_B128(src, &Ks[key][d8]);
    }
    // stage V tile transposed into Vt[dim][key]
    #pragma unroll
    for (int c = tid; c < 256; c += 128) {
      int key = c >> 3, d8 = (c & 7) << 3;
      const __bf16* src = vb + (rowbase + kbase + key) * DD + hb + d8;
      union { uint4 q; __bf16 hh[8]; } u;
      u.q = *reinterpret_cast<const uint4*>(src);
      #pragma unroll
      for (int i = 0; i < 8; ++i) Vt[d8 + i][key] = u.hh[i];
    }
    WAIT_ASYNC();
    __syncthreads();

    // K^T B-frags from LDS: lane col = key, K-axis = head dim (contiguous)
    v16bf k00 = ld_frag(&Ks[ll][lh * 8],          &Ks[ll][16 + lh * 8]);
    v16bf k01 = ld_frag(&Ks[ll][32 + lh * 8],     &Ks[ll][48 + lh * 8]);
    v16bf k10 = ld_frag(&Ks[16 + ll][lh * 8],     &Ks[16 + ll][16 + lh * 8]);
    v16bf k11 = ld_frag(&Ks[16 + ll][32 + lh * 8],&Ks[16 + ll][48 + lh * 8]);

    v8f c0 = {}, c1 = {};
    c0 = wmma_bf16(a0, k00, c0);
    c0 = wmma_bf16(a1, k01, c0);
    c1 = wmma_bf16(a0, k10, c1);
    c1 = wmma_bf16(a1, k11, c1);

    int mv0 = mrow[kbase + ll], mv1 = mrow[kbase + 16 + ll];
    float alpha[8];
    #pragma unroll
    for (int r = 0; r < 8; ++r) {
      float s0 = c0[r] * 0.125f; if (!mv0) s0 = -1e9f;   // 1/sqrt(64)
      float s1 = c1[r] * 0.125f; if (!mv1) s1 = -1e9f;
      float tm = rmax16(fmaxf(s0, s1));
      float mn = fmaxf(rm[r], tm);
      alpha[r] = __expf(rm[r] - mn);
      rm[r] = mn;
      float p0 = __expf(s0 - mn), p1 = __expf(s1 - mn);
      c0[r] = p0; c1[r] = p1;
      rl[r] = rl[r] * alpha[r] + rsum16(p0 + p1);
      #pragma unroll
      for (int t = 0; t < 4; ++t) acc[t][r] *= alpha[r];
    }
    // stage P (C layout) -> LDS -> reload in A-frag layout (per-wave region)
    #pragma unroll
    for (int r = 0; r < 8; ++r) {
      int mr = r + lh * 8;
      pw[mr * 40 + ll]      = f2bf(c0[r]);
      pw[mr * 40 + 16 + ll] = f2bf(c1[r]);
    }
    v16bf pa = ld_frag(&pw[ll * 40 + lh * 8], &pw[ll * 40 + 16 + lh * 8]);

    // P @ V : B-frags from transposed V tile, contiguous along keys
    #pragma unroll
    for (int t = 0; t < 4; ++t) {
      v16bf vf = ld_frag(&Vt[t * 16 + ll][lh * 8], &Vt[t * 16 + ll][16 + lh * 8]);
      acc[t] = wmma_bf16(pa, vf, acc[t]);
    }
  }
  // normalize + write bf16 attention output
  #pragma unroll
  for (int t = 0; t < 4; ++t) {
    #pragma unroll
    for (int r = 0; r < 8; ++r) {
      float o = acc[t][r] / rl[r];
      ob[(rowbase + q0 + r + lh * 8) * DD + hb + t * 16 + ll] = f2bf(o);
    }
  }
}

// ---- workspace layout (bytes) ----
#define OFF_X  ((size_t)0)            // f32 [M,D]  16MB
#define OFF_H  ((size_t)16  << 20)    // bf16 [M,D]  8MB
#define OFF_Q  ((size_t)24  << 20)
#define OFF_K  ((size_t)32  << 20)
#define OFF_V  ((size_t)40  << 20)
#define OFF_O  ((size_t)48  << 20)
#define OFF_T  ((size_t)56  << 20)    // bf16 [M,F] 32MB
#define OFF_WQ ((size_t)88  << 20)    // bf16 [L,D,D] 6MB each
#define OFF_WK ((size_t)94  << 20)
#define OFF_WV ((size_t)100 << 20)
#define OFF_WO ((size_t)106 << 20)
#define OFF_W1 ((size_t)112 << 20)    // bf16 [L,D,F] 24MB
#define OFF_W2 ((size_t)136 << 20)    // bf16 [L,F,D] 24MB

extern "C" void kernel_launch(void* const* d_in, const int* in_sizes, int n_in,
                              void* d_out, int out_size, void* d_ws, size_t ws_size,
                              hipStream_t stream) {
  const int*   tokens = (const int*)d_in[0];
  const int*   mask   = (const int*)d_in[1];
  const float* emb    = (const float*)d_in[2];
  const float* pe     = (const float*)d_in[3];
  const float* Wq     = (const float*)d_in[4];
  const float* bq     = (const float*)d_in[5];
  const float* Wk     = (const float*)d_in[6];
  const float* bk     = (const float*)d_in[7];
  const float* Wv     = (const float*)d_in[8];
  const float* bv     = (const float*)d_in[9];
  const float* Wo     = (const float*)d_in[10];
  const float* bo     = (const float*)d_in[11];
  const float* w1     = (const float*)d_in[12];
  const float* b1     = (const float*)d_in[13];
  const float* w2     = (const float*)d_in[14];
  const float* b2     = (const float*)d_in[15];
  const float* ln_a   = (const float*)d_in[16];
  const float* ln_b   = (const float*)d_in[17];
  const float* fa     = (const float*)d_in[18];
  const float* fb     = (const float*)d_in[19];

  char* ws = (char*)d_ws;
  float*  x   = (float*)(ws + OFF_X);
  __bf16* hbf = (__bf16*)(ws + OFF_H);
  __bf16* qbf = (__bf16*)(ws + OFF_Q);
  __bf16* kbf = (__bf16*)(ws + OFF_K);
  __bf16* vbf = (__bf16*)(ws + OFF_V);
  __bf16* obf = (__bf16*)(ws + OFF_O);
  __bf16* tbf = (__bf16*)(ws + OFF_T);
  __bf16* wqb = (__bf16*)(ws + OFF_WQ);
  __bf16* wkb = (__bf16*)(ws + OFF_WK);
  __bf16* wvb = (__bf16*)(ws + OFF_WV);
  __bf16* wob = (__bf16*)(ws + OFF_WO);
  __bf16* w1b = (__bf16*)(ws + OFF_W1);
  __bf16* w2b = (__bf16*)(ws + OFF_W2);

  // weights -> bf16
  {
    int n = LL * DD * DD;
    int g = (n + 255) / 256;
    f32_to_bf16_k<<<g, 256, 0, stream>>>(Wq, wqb, n);
    f32_to_bf16_k<<<g, 256, 0, stream>>>(Wk, wkb, n);
    f32_to_bf16_k<<<g, 256, 0, stream>>>(Wv, wvb, n);
    f32_to_bf16_k<<<g, 256, 0, stream>>>(Wo, wob, n);
    int nf = LL * DD * FF;
    int gf = (nf + 255) / 256;
    f32_to_bf16_k<<<gf, 256, 0, stream>>>(w1, w1b, nf);
    f32_to_bf16_k<<<gf, 256, 0, stream>>>(w2, w2b, nf);
  }

  embed_k<<<MM, 256, 0, stream>>>(tokens, emb, pe, x);

  dim3 gD(MM / 128, DD / 128);   // N = D
  dim3 gF(MM / 128, FF / 128);   // N = F
  dim3 gA(SS / 64, HH, BB);

  for (int l = 0; l < LL; ++l) {
    // attention sublayer
    ln_k<0><<<MM, 256, 0, stream>>>(x, ln_a + (size_t)(l * 2 + 0) * DD,
                                    ln_b + (size_t)(l * 2 + 0) * DD, hbf, nullptr);
    gemm_bf16_k<0><<<gD, 256, 0, stream>>>(hbf, wqb + (size_t)l * DD * DD, bq + (size_t)l * DD,
                                           qbf, nullptr, MM, DD, DD);
    gemm_bf16_k<0><<<gD, 256, 0, stream>>>(hbf, wkb + (size_t)l * DD * DD, bk + (size_t)l * DD,
                                           kbf, nullptr, MM, DD, DD);
    gemm_bf16_k<0><<<gD, 256, 0, stream>>>(hbf, wvb + (size_t)l * DD * DD, bv + (size_t)l * DD,
                                           vbf, nullptr, MM, DD, DD);
    attn_k<<<gA, 128, 0, stream>>>(qbf, kbf, vbf, mask, obf);
    gemm_bf16_k<2><<<gD, 256, 0, stream>>>(obf, wob + (size_t)l * DD * DD, bo + (size_t)l * DD,
                                           nullptr, x, MM, DD, DD);
    // FFN sublayer
    ln_k<0><<<MM, 256, 0, stream>>>(x, ln_a + (size_t)(l * 2 + 1) * DD,
                                    ln_b + (size_t)(l * 2 + 1) * DD, hbf, nullptr);
    gemm_bf16_k<1><<<gF, 256, 0, stream>>>(hbf, w1b + (size_t)l * DD * FF, b1 + (size_t)l * FF,
                                           tbf, nullptr, MM, FF, DD);
    gemm_bf16_k<2><<<gD, 256, 0, stream>>>(tbf, w2b + (size_t)l * FF * DD, b2 + (size_t)l * DD,
                                           nullptr, x, MM, DD, FF);
  }
  ln_k<1><<<MM, 256, 0, stream>>>(x, fa, fb, nullptr, (float*)d_out);
}
